// INRRandomGraph3D_44341242364139
// MI455X (gfx1250) — matrix operand order repaired and checked by reading, not compiled, not run on hardware
//
#include <hip/hip_runtime.h>

typedef _Float16 half16 __attribute__((ext_vector_type(16)));
typedef _Float16 half8  __attribute__((ext_vector_type(8)));
typedef float    floatx8 __attribute__((ext_vector_type(8)));
typedef float    floatx4 __attribute__((ext_vector_type(4)));

#define LW    64
#define NDIM  16
#define WAVES 4
#define NMAT  12   // W1 + 11 graph matrices

constexpr int WMAT_HALFS = NMAT * LW * LW;        // 49152 halfs (96 KB)
constexpr int WN_HALFS   = LW * NDIM;             // 1024 halfs (2 KB)  W_noise^T
constexpr int WO_HALFS   = 16 * LW;               // 1024 halfs (2 KB)  W_out^T padded 3->16
constexpr int ACT_HALFS  = WAVES * 6 * 16 * LW;   // 24576 halfs (48 KB): 6 slots/wave
constexpr int FB_FLOATS  = 1092;                  // biases + rank-1 weights
constexpr size_t SMEM_BYTES =
    (size_t)(WMAT_HALFS + WN_HALFS + WO_HALFS + ACT_HALFS) * 2 + (size_t)FB_FLOATS * 4;

// fb layout (floats): [0,64)=b1  [64,768)=bg(11x64)  [768,832)=b_noise
// [832,896)=W_x [896,960)=W_y [960,1024)=W_z [1024,1088)=W_r  [1088,1091)=b_out

// ---------- branch-free native transcendentals (CDNA5 TRANS unit) ----------
__device__ __forceinline__ float fexp(float x) {
#if __has_builtin(__builtin_amdgcn_exp2f)
    return __builtin_amdgcn_exp2f(x * 1.4426950408889634f);   // v_exp_f32
#else
    return __expf(x);
#endif
}
__device__ __forceinline__ float frcp(float x) {
#if __has_builtin(__builtin_amdgcn_rcpf)
    return __builtin_amdgcn_rcpf(x);                          // v_rcp_f32
#else
    return 1.f / x;
#endif
}
__device__ __forceinline__ float fln(float x) {               // ln(x), x>0
#if __has_builtin(__builtin_amdgcn_logf)
    return __builtin_amdgcn_logf(x) * 0.6931471805599453f;    // v_log_f32
#else
    return __logf(x);
#endif
}
__device__ __forceinline__ float fsin(float x) {
#if __has_builtin(__builtin_amdgcn_sinf)
    return __builtin_amdgcn_sinf(x * 0.15915494309189535f);   // v_sin_f32 (revolutions)
#else
    return __sinf(x);
#endif
}
__device__ __forceinline__ float ftanh(float x) {
#if __has_builtin(__builtin_amdgcn_tanhf)
    return __builtin_amdgcn_tanhf(x);                         // v_tanh_f32 (gfx1250)
#else
    const float e = fexp(2.f * x);
    return 1.f - 2.f * frcp(e + 1.f);
#endif
}
__device__ __forceinline__ float fsigmoid(float x) { return frcp(1.f + fexp(-x)); }

__device__ __forceinline__ float apply_act(int A, float x) {
    switch (A) {
    case 0:  return ftanh(x);                                       // tanh
    case 1:  return x > 0.f ? x : (fexp(x) - 1.f);                  // elu
    case 2:  return fmaxf(x, 0.f) + fln(1.f + fexp(-fabsf(x)));     // softplus
    case 3:  return fsin(x);                                        // sin
    case 4:  return fexp(-0.5f * x * x);                            // gauss
    default: return fsigmoid(x);                                    // sigmoid
    }
}

// A fragment (16x32 f16): lane m=lo, j<8 -> K=kt*32+hi*8+j, j>=8 -> +16
__device__ __forceinline__ half16 load_a_frag(const _Float16* t, int m, int hi, int kt) {
    const half8* p = (const half8*)(t + m * LW + kt * 32 + hi * 8);
    half8 a = p[0];
    half8 b = p[2]; // +16 halfs
    half16 r;
#pragma unroll
    for (int j = 0; j < 8; ++j) { r[j] = a[j]; r[8 + j] = b[j]; }
    return r;
}

// B fragment (32x16 f16): lane n, j<8 -> k = kt*32 + hi*8 + j, j>=8 -> +16 ; weights stored Wt[n][k]
__device__ __forceinline__ half16 load_b_frag(const _Float16* wt, int n, int kt, int hi) {
    const half8* p = (const half8*)(wt + n * LW + kt * 32 + hi * 8);
    half8 a = p[0];
    half8 b = p[2]; // +16 halfs
    half16 r;
#pragma unroll
    for (int j = 0; j < 8; ++j) { r[j] = a[j]; r[8 + j] = b[j]; }
    return r;
}

// One graph node: dst = act( (p0 [+p1] [+p2]) @ W + bias ), 16x64 tile per wave
__device__ __forceinline__ void mm_node(const _Float16* __restrict__ wt,
                                        const float* __restrict__ bias,
                                        _Float16* __restrict__ dst,
                                        const _Float16* p0, const _Float16* p1,
                                        const _Float16* p2,
                                        int lo, int hi, int actId) {
    half16 a0 = load_a_frag(p0, lo, hi, 0);
    half16 a1 = load_a_frag(p0, lo, hi, 1);
    if (p1) { a0 = a0 + load_a_frag(p1, lo, hi, 0); a1 = a1 + load_a_frag(p1, lo, hi, 1); }
    if (p2) { a0 = a0 + load_a_frag(p2, lo, hi, 0); a1 = a1 + load_a_frag(p2, lo, hi, 1); }
#pragma unroll
    for (int u = 0; u < 4; ++u) {
        const int n = u * 16 + lo;
        floatx8 c;
#pragma unroll
        for (int v = 0; v < 8; ++v) c[v] = 0.f;
        half16 b0 = load_b_frag(wt, n, 0, hi);
        c = __builtin_amdgcn_wmma_f32_16x16x32_f16(false, a0, false, b0, (short)0, c, false, false);
        half16 b1 = load_b_frag(wt, n, 1, hi);
        c = __builtin_amdgcn_wmma_f32_16x16x32_f16(false, a1, false, b1, (short)0, c, false, false);
        const float bv = bias[n];
#pragma unroll
        for (int v = 0; v < 8; ++v)
            dst[(hi * 8 + v) * LW + n] = (_Float16)apply_act(actId, c[v] + bv);
    }
}

__global__ void __launch_bounds__(WAVES * 32)
inr_graph_kernel(const float* __restrict__ X, const float* __restrict__ Y,
                 const float* __restrict__ Z, const float* __restrict__ R,
                 const float* __restrict__ NZ,
                 const float* __restrict__ Wn, const float* __restrict__ bnse,
                 const float* __restrict__ Wx, const float* __restrict__ Wy,
                 const float* __restrict__ Wz, const float* __restrict__ Wr,
                 const float* __restrict__ W1, const float* __restrict__ b1,
                 const float* __restrict__ Wg, const float* __restrict__ bg,
                 const float* __restrict__ Wo, const float* __restrict__ bo,
                 float* __restrict__ out, int nRows) {
    extern __shared__ __align__(16) char smem[];
    _Float16* sw   = (_Float16*)smem;              // 12 x 64x64 f16, transposed [n][k]
    _Float16* swn  = sw + WMAT_HALFS;              // W_noise transposed [n][k16]
    _Float16* swo  = swn + WN_HALFS;               // W_out transposed, N padded to 16
    _Float16* sact = swo + WO_HALFS;               // per-wave activation slots
    float* fb = (float*)(smem +
                 (size_t)(WMAT_HALFS + WN_HALFS + WO_HALFS + ACT_HALFS) * 2);

    const int tid = threadIdx.x;

    // ---- Phase 1: stage weights into LDS (f16, transposed for B-fragment loads)
    for (int idx = tid; idx < NMAT * LW * LW; idx += blockDim.x) {
        int mat = idx >> 12;
        int e = idx & 4095;
        int k = e >> 6;
        int n = e & 63;
        const float* src = (mat == 0) ? W1 : (Wg + (size_t)(mat - 1) * LW * LW);
        sw[mat * LW * LW + n * LW + k] = (_Float16)src[k * LW + n];
    }
    for (int idx = tid; idx < NDIM * LW; idx += blockDim.x) {
        int k = idx >> 6;
        int n = idx & 63;
        swn[n * NDIM + k] = (_Float16)Wn[idx];
    }
    for (int idx = tid; idx < 16 * LW; idx += blockDim.x) {
        int n = idx >> 6;                 // padded output column 0..15
        int k = idx & 63;
        swo[n * LW + k] = (n < 3) ? (_Float16)Wo[k * 3 + n] : (_Float16)0.f;
    }
    for (int i = tid; i < FB_FLOATS; i += blockDim.x) {
        float v = 0.f;
        if      (i < 64)   v = b1[i];
        else if (i < 768)  v = bg[i - 64];
        else if (i < 832)  v = bnse[i - 768];
        else if (i < 896)  v = Wx[i - 832];
        else if (i < 960)  v = Wy[i - 896];
        else if (i < 1024) v = Wz[i - 960];
        else if (i < 1088) v = Wr[i - 1024];
        else if (i < 1091) v = bo[i - 1088];
        fb[i] = v;
    }
    __syncthreads();

    const int wave = tid >> 5;
    const int lane = tid & 31;
    const int lo   = lane & 15;
    const int hi   = lane >> 4;

    _Float16* aw = sact + wave * (6 * 16 * LW);
    _Float16* S0 = aw;
    _Float16* S1 = aw + 1024;
    _Float16* S2 = aw + 2048;
    _Float16* S3 = aw + 3072;
    _Float16* S4 = aw + 4096;
    _Float16* S5 = aw + 5120;

    const int NT = nRows >> 4;
    for (int tile = blockIdx.x * WAVES + wave; tile < NT; tile += gridDim.x * WAVES) {
        const int base = tile << 4;

        // rank-1 branch row scalars for rows base+hi*8 .. +7
        floatx4 x0v = *(const floatx4*)(X + base + hi * 8);
        floatx4 x1v = *(const floatx4*)(X + base + hi * 8 + 4);
        floatx4 y0v = *(const floatx4*)(Y + base + hi * 8);
        floatx4 y1v = *(const floatx4*)(Y + base + hi * 8 + 4);
        floatx4 z0v = *(const floatx4*)(Z + base + hi * 8);
        floatx4 z1v = *(const floatx4*)(Z + base + hi * 8 + 4);
        floatx4 r0v = *(const floatx4*)(R + base + hi * 8);
        floatx4 r1v = *(const floatx4*)(R + base + hi * 8 + 4);

        // ---- Stage A: noise matmul (K padded 16->32) + rank-1 branches -> f0 in S0
        half16 an;
#pragma unroll
        for (int j = 0; j < 16; ++j) an[j] = (_Float16)0.f;
        {
            const floatx4* pn = (const floatx4*)(NZ + (size_t)(base + lo) * NDIM + hi * 8);
            floatx4 n0 = pn[0], n1 = pn[1];
#pragma unroll
            for (int j = 0; j < 4; ++j) { an[j] = (_Float16)n0[j]; an[4 + j] = (_Float16)n1[j]; }
        }
#pragma unroll
        for (int u = 0; u < 4; ++u) {
            const int n = u * 16 + lo;
            half16 bn16;
#pragma unroll
            for (int j = 0; j < 16; ++j) bn16[j] = (_Float16)0.f;
            { // elements 8..15 hold K=16..31 of B: zero padding
                const half8* q = (const half8*)(swn + n * NDIM + hi * 8);
                half8 q0 = q[0];
#pragma unroll
                for (int j = 0; j < 8; ++j) { bn16[j] = q0[j]; }
            }
            floatx8 c;
#pragma unroll
            for (int v = 0; v < 8; ++v) c[v] = 0.f;
            c = __builtin_amdgcn_wmma_f32_16x16x32_f16(false, an, false, bn16, (short)0, c, false, false);

            const float bnv = fb[768 + n];
            const float wxv = fb[832 + n], wyv = fb[896 + n];
            const float wzv = fb[960 + n], wrv = fb[1024 + n];
#pragma unroll
            for (int v = 0; v < 8; ++v) {
                const float xs = (v < 4) ? x0v[v] : x1v[v - 4];
                const float ys = (v < 4) ? y0v[v] : y1v[v - 4];
                const float zs = (v < 4) ? z0v[v] : z1v[v - 4];
                const float rs = (v < 4) ? r0v[v] : r1v[v - 4];
                const float nzv = ftanh(c[v] + bnv);
                const float zz = zs * wzv;
                const float z_ = zz > 0.f ? zz : (fexp(zz) - 1.f);
                const float rp = rs * wrv;
                const float r_ = fmaxf(rp, 0.f) + fln(1.f + fexp(-fabsf(rp)));
                const float y_ = ftanh(ys * wyv);
                const float xx = xs * wxv;
                const float x_ = fexp(-0.5f * xx * xx);
                const float f0 = fsin(z_ + x_ + y_ + r_ + nzv);
                S0[(hi * 8 + v) * LW + n] = (_Float16)f0;
            }
        }

        // ---- h0 = tanh(f0 @ W1 + b1); graph nodes with 6-slot liveness reuse
        mm_node(sw,             fb,       S0, S0, nullptr, nullptr, lo, hi, 0); // h0
        mm_node(sw + 1 * 4096,  fb + 64,  S1, S0, nullptr, nullptr, lo, hi, 0); // h1 tanh
        mm_node(sw + 2 * 4096,  fb + 128, S2, S1, nullptr, nullptr, lo, hi, 1); // h2 elu
        mm_node(sw + 3 * 4096,  fb + 192, S3, S2, S0,      nullptr, lo, hi, 2); // h3 softplus
        mm_node(sw + 4 * 4096,  fb + 256, S4, S3, S1,      S0,      lo, hi, 3); // h4 sin
        mm_node(sw + 5 * 4096,  fb + 320, S5, S4, S2,      nullptr, lo, hi, 4); // h5 gauss
        mm_node(sw + 6 * 4096,  fb + 384, S2, S5, S3,      nullptr, lo, hi, 5); // h6 sigmoid
        mm_node(sw + 7 * 4096,  fb + 448, S3, S2, S4,      nullptr, lo, hi, 0); // h7 tanh
        mm_node(sw + 8 * 4096,  fb + 512, S4, S3, S5,      S0,      lo, hi, 1); // h8 elu
        mm_node(sw + 9 * 4096,  fb + 576, S5, S4, S2,      nullptr, lo, hi, 2); // h9 softplus
        mm_node(sw + 10 * 4096, fb + 640, S0, S5, S3,      nullptr, lo, hi, 3); // h10 sin
        mm_node(sw + 11 * 4096, fb + 704, S1, S0, S4,      nullptr, lo, hi, 4); // h11 gauss

        // ---- out = sigmoid(h11 @ W_out + b_out) via WMMA (N padded 3->16)
        {
            half16 a0 = load_a_frag(S1, lo, hi, 0);
            half16 a1 = load_a_frag(S1, lo, hi, 1);
            floatx8 c;
#pragma unroll
            for (int v = 0; v < 8; ++v) c[v] = 0.f;
            half16 b0 = load_b_frag(swo, lo, 0, hi);
            c = __builtin_amdgcn_wmma_f32_16x16x32_f16(false, a0, false, b0, (short)0, c, false, false);
            half16 b1 = load_b_frag(swo, lo, 1, hi);
            c = __builtin_amdgcn_wmma_f32_16x16x32_f16(false, a1, false, b1, (short)0, c, false, false);
            if (lo < 3) { // columns 3..15 are padding
                const float bov = fb[1088 + lo];
#pragma unroll
                for (int v = 0; v < 8; ++v) {
                    const int m = hi * 8 + v;
                    out[(size_t)(base + m) * 3 + lo] = fsigmoid(c[v] + bov);
                }
            }
        }
    }
}

extern "C" void kernel_launch(void* const* d_in, const int* in_sizes, int n_in,
                              void* d_out, int out_size, void* d_ws, size_t ws_size,
                              hipStream_t stream) {
    (void)n_in; (void)out_size; (void)d_ws; (void)ws_size;
    const float* X   = (const float*)d_in[0];
    const float* Y   = (const float*)d_in[1];
    const float* Z   = (const float*)d_in[2];
    const float* R   = (const float*)d_in[3];
    const float* NZ  = (const float*)d_in[4];
    const float* Wn  = (const float*)d_in[5];
    const float* bn  = (const float*)d_in[6];
    const float* Wx  = (const float*)d_in[7];
    const float* Wy  = (const float*)d_in[8];
    const float* Wz  = (const float*)d_in[9];
    const float* Wr  = (const float*)d_in[10];
    const float* W1  = (const float*)d_in[11];
    const float* b1  = (const float*)d_in[12];
    const float* Wg  = (const float*)d_in[13];
    const float* bgp = (const float*)d_in[14];
    const float* Wo  = (const float*)d_in[15];
    const float* bo  = (const float*)d_in[16];
    float* out = (float*)d_out;
    const int nRows = in_sizes[0];

    static_assert(SMEM_BYTES < 320u * 1024u, "LDS budget");
    (void)hipFuncSetAttribute(reinterpret_cast<const void*>(inr_graph_kernel),
                              hipFuncAttributeMaxDynamicSharedMemorySize,
                              (int)SMEM_BYTES);
    dim3 grid(1024), block(WAVES * 32);
    inr_graph_kernel<<<grid, block, (uint32_t)SMEM_BYTES, stream>>>(
        X, Y, Z, R, NZ, Wn, bn, Wx, Wy, Wz, Wr, W1, b1, Wg, bgp, Wo, bo, out, nRows);
}
